// AwkwardRNNDoubleJagged_1176821039417
// MI455X (gfx1250) — compile-verified, matrix operand order checked by reading, not run on hardware
//
#include <hip/hip_runtime.h>
#include <hip/hip_bf16.h>
#include <stdint.h>

// ---------------------------------------------------------------------------
// Sequential LSTM (H=2048, P=64, T=32) for MI455X / gfx1250.
// Recurrent matvec W_hh@h done with v_wmma_f32_16x16x32_bf16, weights packed
// once into WMMA A-fragment layout in bf16 (32MB, L2-resident).
// One fused kernel per timestep; launch order provides the sequential sync.
// ---------------------------------------------------------------------------

#define HID   2048
#define G4H   8192        // 4*HID gate rows
#define NPART 64
#define NT    32
#define KCH   64          // 2048 / 32 k-chunks
#define HTILES 128        // 2048 / 16 hidden tiles
#define NWAVES 8          // waves per step-block (split-K)

typedef __attribute__((ext_vector_type(16))) __bf16 v16bf;
typedef __attribute__((ext_vector_type(8)))  __bf16 v8bf;
typedef __attribute__((ext_vector_type(8)))  float  v8f;

__device__ __forceinline__ unsigned short f32_to_bf16_rne(float f) {
    unsigned u = __float_as_uint(f);
    u += 0x7FFFu + ((u >> 16) & 1u);
    return (unsigned short)(u >> 16);
}
__device__ __forceinline__ float sigmoidf_fast(float x) {
    return 1.0f / (1.0f + __expf(-x));
}

// ---------------------------------------------------------------------------
// Pack W_hh (fp32 row-major 8192x2048) into bf16 WMMA A-fragments.
// Fragment block = (row_tile rt 0..511, k_chunk kc 0..63): 32 lanes x 16 bf16.
// Lane l element j holds W[rt*16 + (l&15), kc*32 + koff] where
//   lanes 0-15  : koff in {0..7, 16..23}
//   lanes 16-31 : koff in {8..15, 24..31}        (ISA 16-bit A 16x32 layout)
// Output index == linear tid -> coalesced stores.
// ---------------------------------------------------------------------------
__global__ void pack_weights(const float* __restrict__ Whh,
                             unsigned short* __restrict__ Wp) {
    int tid  = blockIdx.x * blockDim.x + threadIdx.x;   // < 16,777,216
    int j    = tid & 15;
    int lane = (tid >> 4) & 31;
    int blk  = tid >> 9;
    int kc   = blk & (KCH - 1);
    int rt   = blk >> 6;                                // 0..511
    int row  = rt * 16 + (lane & 15);
    int jj   = (j < 8) ? j : j + 8;
    int koff = (lane < 16) ? jj : jj + 8;
    int k    = kc * 32 + koff;
    Wp[tid]  = f32_to_bf16_rne(Whh[(long)row * HID + k]);
}

// bias sum + zero initial state (h, c, both hbf ping-pong buffers)
__global__ void setup_state(const float* __restrict__ b_ih,
                            const float* __restrict__ b_hh,
                            float* __restrict__ bsum,
                            float* __restrict__ h, float* __restrict__ c,
                            unsigned short* __restrict__ hbf0,
                            unsigned short* __restrict__ hbf1) {
    int tid = blockIdx.x * blockDim.x + threadIdx.x;    // < 8192
    bsum[tid] = b_ih[tid] + b_hh[tid];
    if (tid < HID) {
        h[tid] = 0.0f; c[tid] = 0.0f; hbf0[tid] = 0; hbf1[tid] = 0;
    }
}

// Particle carry: h0 = [h_prev[1024:], 0], c0 = [c_prev[1024:], 0].
// Single block so the read->write swap is safe with one __syncthreads.
__global__ void particle_shift(float* __restrict__ h, float* __restrict__ c,
                               unsigned short* __restrict__ hbf0) {
    int tid  = threadIdx.x;                 // 1024 threads
    float hv = h[1024 + tid];
    float cv = c[1024 + tid];
    __syncthreads();
    h[tid] = hv;         c[tid] = cv;         hbf0[tid] = f32_to_bf16_rne(hv);
    h[1024 + tid] = 0.f; c[1024 + tid] = 0.f; hbf0[1024 + tid] = 0;
}

// ---------------------------------------------------------------------------
// One LSTM timestep. Block = 16 hidden units (ht). 8 waves split K (8 chunks
// of 32 each). Each wave does 4 gate tiles x 8 k-chunks = 32 WMMAs, then the
// split-K partials are reduced in LDS and 16 threads run the pointwise cell.
// ---------------------------------------------------------------------------
__global__ __launch_bounds__(256) void lstm_step(
        const unsigned short* __restrict__ Wp,
        const unsigned short* __restrict__ hbf_in,
        unsigned short* __restrict__ hbf_out,
        float* __restrict__ h, float* __restrict__ c,
        const float* __restrict__ bsum, const float* __restrict__ w_in,
        const float* __restrict__ event, const int* __restrict__ lengths,
        int p, int t) {
    const int ht = blockIdx.x;              // 0..127
    if (t >= lengths[p]) {                  // inactive step: keep state, keep
        if (threadIdx.x < 16) {             // ping-pong buffers consistent
            int hid = ht * 16 + (int)threadIdx.x;
            hbf_out[hid] = hbf_in[hid];
        }
        return;
    }
    const int lane = threadIdx.x & 31;
    const int wave = threadIdx.x >> 5;
    const int halfoff = (lane & 16) ? 8 : 0;

    const __bf16* hb = (const __bf16*)hbf_in;
    const __bf16* W  = (const __bf16*)Wp;

    v8f acc0 = {0,0,0,0,0,0,0,0};
    v8f acc1 = {0,0,0,0,0,0,0,0};
    v8f acc2 = {0,0,0,0,0,0,0,0};
    v8f acc3 = {0,0,0,0,0,0,0,0};

    #pragma unroll
    for (int kcl = 0; kcl < 8; ++kcl) {
        const int kc = wave * 8 + kcl;
        const int k0 = kc * 32;
        // B fragment: h chunk replicated across all 16 N columns.
        // lanes 0-15 need K {0..7,16..23}; lanes 16-31 need K {8..15,24..31}.
        v8bf blo = *(const v8bf*)(hb + k0 + halfoff);
        v8bf bhi = *(const v8bf*)(hb + k0 + 16 + halfoff);
        v16bf B;
        #pragma unroll
        for (int i = 0; i < 8; ++i) { B[i] = blo[i]; B[8 + i] = bhi[i]; }

        const long base = (long)kc * 512 + (long)lane * 16;
        const long gstep = (long)HTILES * KCH * 512;     // next gate's tiles
        const long toff  = (long)ht * KCH * 512;

        v16bf A0 = *(const v16bf*)(W + 0 * gstep + toff + base);
        acc0 = __builtin_amdgcn_wmma_f32_16x16x32_bf16(false, A0, false, B,
                                                       (short)0, acc0, false, false);
        v16bf A1 = *(const v16bf*)(W + 1 * gstep + toff + base);
        acc1 = __builtin_amdgcn_wmma_f32_16x16x32_bf16(false, A1, false, B,
                                                       (short)0, acc1, false, false);
        v16bf A2 = *(const v16bf*)(W + 2 * gstep + toff + base);
        acc2 = __builtin_amdgcn_wmma_f32_16x16x32_bf16(false, A2, false, B,
                                                       (short)0, acc2, false, false);
        v16bf A3 = *(const v16bf*)(W + 3 * gstep + toff + base);
        acc3 = __builtin_amdgcn_wmma_f32_16x16x32_bf16(false, A3, false, B,
                                                       (short)0, acc3, false, false);
    }

    // D layout: VGPR r = M row r (lanes 0-15) / row 8+r (lanes 16-31); all N
    // columns identical, so lanes 0 and 16 carry the full 16-row result.
    __shared__ float red[NWAVES][4][16];
    if (lane == 0 || lane == 16) {
        const int hh = lane >> 4;
        #pragma unroll
        for (int r = 0; r < 8; ++r) {
            red[wave][0][hh * 8 + r] = acc0[r];
            red[wave][1][hh * 8 + r] = acc1[r];
            red[wave][2][hh * 8 + r] = acc2[r];
            red[wave][3][hh * 8 + r] = acc3[r];
        }
    }
    __syncthreads();

    if (threadIdx.x < 16) {
        const int j = (int)threadIdx.x;
        float s0 = 0.f, s1 = 0.f, s2 = 0.f, s3 = 0.f;
        #pragma unroll
        for (int w = 0; w < NWAVES; ++w) {
            s0 += red[w][0][j]; s1 += red[w][1][j];
            s2 += red[w][2][j]; s3 += red[w][3][j];
        }
        const int hid = ht * 16 + j;
        const float x = event[p * NT + t];
        float gi = s0 + w_in[hid]            * x + bsum[hid];
        float gf = s1 + w_in[HID     + hid]  * x + bsum[HID     + hid];
        float gg = s2 + w_in[2 * HID + hid]  * x + bsum[2 * HID + hid];
        float go = s3 + w_in[3 * HID + hid]  * x + bsum[3 * HID + hid];
        float iv = sigmoidf_fast(gi);
        float fv = sigmoidf_fast(gf);
        float gv = tanhf(gg);
        float ov = sigmoidf_fast(go);
        float cn = fv * c[hid] + iv * gv;
        float hn = ov * tanhf(cn);
        c[hid] = cn;
        h[hid] = hn;
        hbf_out[hid] = f32_to_bf16_rne(hn);
    }
}

__global__ void copy_out(const float* __restrict__ h, float* __restrict__ out) {
    int tid = blockIdx.x * blockDim.x + threadIdx.x;
    if (tid < HID) out[tid] = h[tid];
}

// ---------------------------------------------------------------------------
extern "C" void kernel_launch(void* const* d_in, const int* in_sizes, int n_in,
                              void* d_out, int out_size, void* d_ws, size_t ws_size,
                              hipStream_t stream) {
    const float* event   = (const float*)d_in[0];   // (P, T) fp32
    const int*   lengths = (const int*)  d_in[1];   // (P,)  int32
    const float* W_ih    = (const float*)d_in[2];   // (8192, 1) -> w_in
    const float* W_hh    = (const float*)d_in[3];   // (8192, 2048)
    const float* b_ih    = (const float*)d_in[4];   // (8192,)
    const float* b_hh    = (const float*)d_in[5];   // (8192,)
    float* out = (float*)d_out;                     // 2048 fp32

    uint8_t* ws = (uint8_t*)d_ws;
    size_t off = 0;
    unsigned short* Wp   = (unsigned short*)(ws + off); off += (size_t)G4H * HID * 2; // 32MB
    float*          bsum = (float*)         (ws + off); off += (size_t)G4H * 4;
    float*          h    = (float*)         (ws + off); off += (size_t)HID * 4;
    float*          c    = (float*)         (ws + off); off += (size_t)HID * 4;
    unsigned short* hbf0 = (unsigned short*)(ws + off); off += (size_t)HID * 2;
    unsigned short* hbf1 = (unsigned short*)(ws + off); off += (size_t)HID * 2;

    // one-time weight repack (bf16 WMMA fragments) + bias/state init
    pack_weights<<<(G4H * HID) / 256, 256, 0, stream>>>(W_hh, Wp);
    setup_state<<<G4H / 256, 256, 0, stream>>>(b_ih, b_hh, bsum, h, c, hbf0, hbf1);

    int s = 0;  // global step parity for hbf ping-pong
    for (int p = 0; p < NPART; ++p) {
        if (p) particle_shift<<<1, 1024, 0, stream>>>(h, c, hbf0); // before even step
        for (int t = 0; t < NT; ++t) {
            unsigned short* hin  = (s & 1) ? hbf1 : hbf0;
            unsigned short* hout = (s & 1) ? hbf0 : hbf1;
            lstm_step<<<HTILES, 256, 0, stream>>>(Wp, hin, hout, h, c, bsum,
                                                  W_ih, event, lengths, p, t);
            ++s;
        }
    }
    copy_out<<<8, 256, 0, stream>>>(h, out);
}
